// ResidualBlock_44581760532912
// MI455X (gfx1250) — matrix operand order repaired and checked by reading, not compiled, not run on hardware
//
#include <hip/hip_runtime.h>

// -------------------------------------------------------------------------
// MI455X (gfx1250) submanifold sparse-conv residual block.
// Compute-bound (~400 GFLOP vs ~13us of HBM traffic) -> bf16 WMMA
// (v_wmma_f32_16x16x32_bf16, f32 accumulate), double-buffered LDS gather
// tiles fed by GLOBAL_LOAD_ASYNC_TO_LDS_B128 (ASYNCcnt-paced, no VGPR
// round trip), weights pre-packed into the ISA per-lane B-fragment layout.
// -------------------------------------------------------------------------

#define G_DIM  160
#define NVOX   300000
#define C_IN   64
#define C_OUT  128
#define BN_EPS 1e-4f

typedef __attribute__((ext_vector_type(16))) __bf16 v16bf;
typedef __attribute__((ext_vector_type(8)))  __bf16 v8bf;
typedef __attribute__((ext_vector_type(8)))  float  v8f;

// Wait for outstanding async->LDS transfers, then workgroup barrier.
// (s_barrier does NOT imply ASYNCcnt completion; must wait explicitly.)
#define WG_SYNC()                                            \
    do {                                                     \
        asm volatile("s_wait_asynccnt 0x0" ::: "memory");    \
        __syncthreads();                                     \
    } while (0)

// ---------------- prep kernels ----------------

__global__ void k_init_grid(int* grid, int total) {
    int i = blockIdx.x * 256 + threadIdx.x;
    if (i < total) grid[i] = -1;
}

__global__ void k_scatter(const int* __restrict__ pos, int* __restrict__ grid, int n) {
    int i = blockIdx.x * 256 + threadIdx.x;
    if (i >= n) return;
    int x = pos[3*i], y = pos[3*i+1], z = pos[3*i+2];
    grid[(x * G_DIM + y) * G_DIM + z] = i;
}

__global__ void k_nbr(const int* __restrict__ pos, const int* __restrict__ grid,
                      int* __restrict__ nbr, int n) {
    int i = blockIdx.x * 256 + threadIdx.x;
    if (i >= n) return;
    int x = pos[3*i], y = pos[3*i+1], z = pos[3*i+2];
    int k = 0;
    for (int dx = -1; dx <= 1; ++dx)
        for (int dy = -1; dy <= 1; ++dy)
            for (int dz = -1; dz <= 1; ++dz, ++k) {
                int nx = x + dx, ny = y + dy, nz = z + dz;
                int v = -1;
                if (nx >= 0 && nx < G_DIM && ny >= 0 && ny < G_DIM &&
                    nz >= 0 && nz < G_DIM)
                    v = grid[(nx * G_DIM + ny) * G_DIM + nz];
                nbr[(size_t)k * NVOX + i] = v;
            }
}

// Pack W [27, cin, 128] f32 -> bf16 B-fragments.
// Packed index t = ((((kk*KT + kt)*8 + nt)*32) + lane)*16 + e
// B element (ISA 16-bit B layout, wave32): K = kt*32 + (lane>=16)*8 + (e&7) + (e>=8)*16,
//                                          N = nt*16 + (lane&15)
__global__ void k_pack_w(const float* __restrict__ W, __bf16* __restrict__ out,
                         int cin, int total) {
    int t = blockIdx.x * 256 + threadIdx.x;
    if (t >= total) return;
    int e    = t & 15;
    int lane = (t >> 4) & 31;
    int nt   = (t >> 9) & 7;
    int rest = t >> 12;
    int ktc  = cin >> 5;
    int kt   = rest % ktc;
    int kk   = rest / ktc;
    int k = kt * 32 + ((lane >> 4) << 3) + (e & 7) + ((e >> 3) << 4);
    int n = nt * 16 + (lane & 15);
    out[t] = (__bf16)W[((size_t)kk * cin + k) * C_OUT + n];
}

// lin_w is [COUT, CIN]; B[k=c][n=o] = lin_w[o*CIN + c]; same packed layout, kk==0.
__global__ void k_pack_lin(const float* __restrict__ linw, __bf16* __restrict__ out,
                           int total) {
    int t = blockIdx.x * 256 + threadIdx.x;
    if (t >= total) return;
    int e    = t & 15;
    int lane = (t >> 4) & 31;
    int nt   = (t >> 9) & 7;
    int kt   = t >> 12;                       // 0..1 (C_IN=64)
    int k = kt * 32 + ((lane >> 4) << 3) + (e & 7) + ((e >> 3) << 4);
    int n = nt * 16 + (lane & 15);
    out[t] = (__bf16)linw[n * C_IN + k];
}

// BN1 + ReLU + bf16 cast; also cast raw feat for the skip GEMM.
__global__ void k_stage0(const float* __restrict__ feat,
                         const float* __restrict__ g, const float* __restrict__ b,
                         const float* __restrict__ mu, const float* __restrict__ var,
                         __bf16* __restrict__ featbf, __bf16* __restrict__ h1bf,
                         int total) {
    int t = blockIdx.x * 256 + threadIdx.x;
    if (t >= total) return;
    int c = t & (C_IN - 1);
    float f = feat[t];
    featbf[t] = (__bf16)f;
    float s = g[c] * rsqrtf(var[c] + BN_EPS);
    float h = (f - mu[c]) * s + b[c];
    h1bf[t] = (__bf16)fmaxf(h, 0.0f);
}

// ---------------- WMMA / async helpers ----------------

__device__ __forceinline__ v8f wmma_bf16(v16bf a, v16bf b, v8f c) {
    return __builtin_amdgcn_wmma_f32_16x16x32_bf16(
        /*neg_a=*/false, a, /*neg_b=*/false, b,
        /*c_mod=*/(short)0, c, /*reuse_a=*/false, /*reuse_b=*/false);
}

// 16B global -> LDS async copy. LDS byte offset = low 32 bits of the flat
// shared address (LDS aperture keeps the offset in addr[31:0], ISA 10.2).
__device__ __forceinline__ void async_g2l_b128(void* lds_dst, const void* gsrc) {
    unsigned int lds_off = (unsigned int)(size_t)lds_dst;
    asm volatile("global_load_async_to_lds_b128 %0, %1, off"
                 :: "v"(lds_off), "v"(gsrc) : "memory");
}

// A-fragment (16x32 bf16, row-major in LDS with leading dim ld):
// lane m = lane&15; K-half kh = 8*(lane>=16);
// VGPRs 0-3 = K kh..kh+7 (one b128), VGPRs 4-7 = K kh+16..kh+23 (one b128).
__device__ __forceinline__ v16bf load_A_frag(const __bf16* base, int ld, int lane) {
    int m  = lane & 15;
    int kh = (lane >> 4) << 3;
    const __bf16* p = base + m * ld + kh;
    v8bf lo = *(const v8bf*)(p);
    v8bf hi = *(const v8bf*)(p + 16);
    return __builtin_shufflevector(lo, hi, 0, 1, 2, 3, 4, 5, 6, 7,
                                           8, 9, 10, 11, 12, 13, 14, 15);
}

// Gather 32 activation rows (bf16, CINT channels) into LDS via async copies;
// idx<0 rows are zero-filled with a regular ds_store.
template <int CINT>
__device__ __forceinline__ void gather_tile(__bf16* __restrict__ dst,
                                            const __bf16* __restrict__ src,
                                            const int* __restrict__ nbr_k, int row0) {
    constexpr int CH = CINT / 8;               // 16B chunks per row
    for (int c = threadIdx.x; c < 32 * CH; c += 256) {
        int j = c / CH, col = (c - j * CH) * 8;
        int idx = nbr_k[row0 + j];
        __bf16* d = dst + j * CINT + col;
        if (idx >= 0) {
            async_g2l_b128(d, src + (size_t)idx * CINT + col);
        } else {
            *(uint4*)d = make_uint4(0u, 0u, 0u, 0u);
        }
    }
}

// ---------------- conv1 (fused skip GEMM + BN2/ReLU epilogue) ----------------

__global__ void __launch_bounds__(256)
k_conv1(const __bf16* __restrict__ h1, const __bf16* __restrict__ fsk,
        const __bf16* __restrict__ B1, const __bf16* __restrict__ Blin,
        const int* __restrict__ nbr,
        const float* __restrict__ g2, const float* __restrict__ b2,
        const float* __restrict__ mu2, const float* __restrict__ var2,
        float* __restrict__ out, __bf16* __restrict__ h2) {
    __shared__ __bf16 sm[2][32][C_IN];   // 8 KB double-buffered gather tile
    __shared__ __bf16 sk[32][C_IN];      // 4 KB skip (feat) tile

    const int row0 = blockIdx.x * 32;    // NVOX % 32 == 0
    const int wave = threadIdx.x >> 5, lane = threadIdx.x & 31;
    const int mt  = wave >> 2;           // 0..1 : M tile
    const int nt0 = (wave & 3) * 2, nt1 = nt0 + 1;  // two N tiles per wave

    for (int c = threadIdx.x; c < 32 * (C_IN / 8); c += 256) {
        int j = c / (C_IN / 8), col = (c - j * (C_IN / 8)) * 8;
        async_g2l_b128(&sk[j][col], fsk + (size_t)(row0 + j) * C_IN + col);
    }
    gather_tile<C_IN>(&sm[0][0][0], h1, nbr, row0);

    v8f acc0 = {}, acc1 = {}, skacc0 = {}, skacc1 = {};
    WG_SYNC();

    // skip GEMM: feat(rows) x lin_w^T
#pragma unroll
    for (int kt = 0; kt < C_IN / 32; ++kt) {
        v16bf a  = load_A_frag(&sk[mt * 16][kt * 32], C_IN, lane);
        v16bf b0 = *(const v16bf*)(Blin + (size_t)((kt * 8 + nt0) * 32 + lane) * 16);
        v16bf b1 = *(const v16bf*)(Blin + (size_t)((kt * 8 + nt1) * 32 + lane) * 16);
        skacc0 = wmma_bf16(a, b0, skacc0);
        skacc1 = wmma_bf16(a, b1, skacc1);
    }

    for (int kk = 0; kk < 27; ++kk) {
        if (kk) WG_SYNC();
        if (kk + 1 < 27)
            gather_tile<C_IN>(&sm[(kk + 1) & 1][0][0], h1,
                              nbr + (size_t)(kk + 1) * NVOX, row0);
        const __bf16* buf = &sm[kk & 1][0][0];
#pragma unroll
        for (int kt = 0; kt < C_IN / 32; ++kt) {
            v16bf a = load_A_frag(buf + mt * 16 * C_IN + kt * 32, C_IN, lane);
            const __bf16* bp = B1 + (size_t)((kk * (C_IN / 32) + kt) * 8) * 32 * 16;
            v16bf b0 = *(const v16bf*)(bp + (size_t)(nt0 * 32 + lane) * 16);
            v16bf b1 = *(const v16bf*)(bp + (size_t)(nt1 * 32 + lane) * 16);
            acc0 = wmma_bf16(a, b0, acc0);
            acc1 = wmma_bf16(a, b1, acc1);
        }
    }

    // C/D layout: VGPR v -> M = v + 8*(lane>=16), N = lane&15
    const int mrow = row0 + mt * 16 + ((lane >> 4) << 3);
    const int nc0 = nt0 * 16 + (lane & 15), nc1 = nt1 * 16 + (lane & 15);
    const float s0 = g2[nc0] * rsqrtf(var2[nc0] + BN_EPS);
    const float s1 = g2[nc1] * rsqrtf(var2[nc1] + BN_EPS);
    const float t0 = b2[nc0] - mu2[nc0] * s0;
    const float t1 = b2[nc1] - mu2[nc1] * s1;
#pragma unroll
    for (int v = 0; v < 8; ++v) {
        size_t r = (size_t)(mrow + v) * C_OUT;
        out[r + nc0] = skacc0[v];
        out[r + nc1] = skacc1[v];
        h2[r + nc0] = (__bf16)fmaxf(acc0[v] * s0 + t0, 0.0f);
        h2[r + nc1] = (__bf16)fmaxf(acc1[v] * s1 + t1, 0.0f);
    }
}

// ---------------- conv2 (accumulates into out) ----------------

__global__ void __launch_bounds__(256)
k_conv2(const __bf16* __restrict__ h2, const __bf16* __restrict__ B2,
        const int* __restrict__ nbr, float* __restrict__ out) {
    __shared__ __bf16 sm[2][32][C_OUT];  // 16 KB double-buffered gather tile

    const int row0 = blockIdx.x * 32;
    const int wave = threadIdx.x >> 5, lane = threadIdx.x & 31;
    const int mt  = wave >> 2;
    const int nt0 = (wave & 3) * 2, nt1 = nt0 + 1;

    gather_tile<C_OUT>(&sm[0][0][0], h2, nbr, row0);
    v8f acc0 = {}, acc1 = {};
    WG_SYNC();

    for (int kk = 0; kk < 27; ++kk) {
        if (kk) WG_SYNC();
        if (kk + 1 < 27)
            gather_tile<C_OUT>(&sm[(kk + 1) & 1][0][0], h2,
                               nbr + (size_t)(kk + 1) * NVOX, row0);
        const __bf16* buf = &sm[kk & 1][0][0];
#pragma unroll
        for (int kt = 0; kt < C_OUT / 32; ++kt) {
            v16bf a = load_A_frag(buf + mt * 16 * C_OUT + kt * 32, C_OUT, lane);
            const __bf16* bp = B2 + (size_t)((kk * (C_OUT / 32) + kt) * 8) * 32 * 16;
            v16bf b0 = *(const v16bf*)(bp + (size_t)(nt0 * 32 + lane) * 16);
            v16bf b1 = *(const v16bf*)(bp + (size_t)(nt1 * 32 + lane) * 16);
            acc0 = wmma_bf16(a, b0, acc0);
            acc1 = wmma_bf16(a, b1, acc1);
        }
    }

    const int mrow = row0 + mt * 16 + ((lane >> 4) << 3);
    const int nc0 = nt0 * 16 + (lane & 15), nc1 = nt1 * 16 + (lane & 15);
#pragma unroll
    for (int v = 0; v < 8; ++v) {
        size_t r = (size_t)(mrow + v) * C_OUT;
        out[r + nc0] += acc0[v];
        out[r + nc1] += acc1[v];
    }
}

// ---------------- host launcher ----------------

extern "C" void kernel_launch(void* const* d_in, const int* in_sizes, int n_in,
                              void* d_out, int out_size, void* d_ws, size_t ws_size,
                              hipStream_t stream) {
    const float* feat  = (const float*)d_in[0];
    const int*   pos   = (const int*)d_in[1];
    const float* lin_w = (const float*)d_in[2];
    const float* bn1g  = (const float*)d_in[3];
    const float* bn1b  = (const float*)d_in[4];
    const float* bn1m  = (const float*)d_in[5];
    const float* bn1v  = (const float*)d_in[6];
    const float* W1    = (const float*)d_in[7];
    const float* bn2g  = (const float*)d_in[8];
    const float* bn2b  = (const float*)d_in[9];
    const float* bn2m  = (const float*)d_in[10];
    const float* bn2v  = (const float*)d_in[11];
    const float* W2    = (const float*)d_in[12];
    float* out = (float*)d_out;

    // workspace layout (~204 MB; all sub-arrays 32B aligned)
    int*    grid   = (int*)d_ws;                                     // 16.38 MB
    int*    nbr    = grid + (size_t)G_DIM * G_DIM * G_DIM;           // 32.40 MB
    __bf16* featbf = (__bf16*)(nbr + 27 * (size_t)NVOX);             // 38.40 MB
    __bf16* h1bf   = featbf + (size_t)NVOX * C_IN;                   // 38.40 MB
    __bf16* h2bf   = h1bf + (size_t)NVOX * C_IN;                     // 76.80 MB
    __bf16* B1     = h2bf + (size_t)NVOX * C_OUT;                    //  0.44 MB
    __bf16* Blin   = B1 + 27 * C_IN * C_OUT;                         //  0.02 MB
    __bf16* B2     = Blin + C_IN * C_OUT;                            //  0.88 MB

    const int g3 = G_DIM * G_DIM * G_DIM;
    k_init_grid<<<(g3 + 255) / 256, 256, 0, stream>>>(grid, g3);
    k_scatter<<<(NVOX + 255) / 256, 256, 0, stream>>>(pos, grid, NVOX);
    k_nbr<<<(NVOX + 255) / 256, 256, 0, stream>>>(pos, grid, nbr, NVOX);

    k_pack_w<<<(27 * C_IN * C_OUT + 255) / 256, 256, 0, stream>>>(
        W1, B1, C_IN, 27 * C_IN * C_OUT);
    k_pack_w<<<(27 * C_OUT * C_OUT + 255) / 256, 256, 0, stream>>>(
        W2, B2, C_OUT, 27 * C_OUT * C_OUT);
    k_pack_lin<<<(C_IN * C_OUT + 255) / 256, 256, 0, stream>>>(
        lin_w, Blin, C_IN * C_OUT);

    k_stage0<<<(NVOX * C_IN + 255) / 256, 256, 0, stream>>>(
        feat, bn1g, bn1b, bn1m, bn1v, featbf, h1bf, NVOX * C_IN);

    k_conv1<<<NVOX / 32, 256, 0, stream>>>(h1bf, featbf, B1, Blin, nbr,
                                           bn2g, bn2b, bn2m, bn2v, out, h2bf);
    k_conv2<<<NVOX / 32, 256, 0, stream>>>(h2bf, B2, nbr, out);
}
